// GRU_SELF_56530359550575
// MI455X (gfx1250) — matrix operand order repaired and checked by reading, not compiled, not run on hardware
//
#include <hip/hip_runtime.h>
#include <hip/hip_bf16.h>
#include <math.h>

#define BDIM 32
#define LDIM 32
#define EDIM 512
#define HDIM 1024
#define NDIM 512
#define VDIM 50257
#define CATD 2560   /* 2*H + E */

#define GBM 32
#define GBN 64
#define GBK 32
#define APAD 56     /* LDS row pitch (ushorts) => 112B: 16B aligned, conflict-free b128 reads */
#define BPAD 56

typedef __bf16 v16bf __attribute__((ext_vector_type(16)));
typedef float  v8f   __attribute__((ext_vector_type(8)));

struct U32x8 { uint4 lo, hi; };

__device__ __forceinline__ unsigned short f2bf(float f) {
  unsigned int u = __float_as_uint(f);
  u += 0x7fffu + ((u >> 16) & 1u);       // round-to-nearest-even
  return (unsigned short)(u >> 16);
}
__device__ __forceinline__ unsigned int f2bf2(float lo, float hi) {
  return (unsigned int)f2bf(lo) | ((unsigned int)f2bf(hi) << 16);
}
__device__ __forceinline__ float sigf(float x) { return 1.0f / (1.0f + expf(-x)); }

// ---------------------------------------------------------------------------
// Generic GEMM: Out[perm(r)] (+)= A(f32, MxK) * B(f32, KxN) (+ bias)
// A,B converted to bf16 while staging into LDS; f32 accumulation via
// v_wmma_f32_16x16x32_bf16. Output row r maps to
// (r % rows_mod)*stride_in + (r / rows_mod)*stride_out.
// Staging uses uniform (SGPR) base + 32-bit per-lane offsets -> saddr form.
// ---------------------------------------------------------------------------
__global__ __launch_bounds__(128) void gemm_bf16_wmma(
    const float* __restrict__ A, int lda,
    const float* __restrict__ B, int ldb,
    const float* __restrict__ bias,
    float* __restrict__ Out,
    int M, int N, int K,
    int rows_mod, long long stride_in, long long stride_out,
    int accum)
{
  __shared__ __align__(16) unsigned short As[GBM * APAD];   // [m][k]
  __shared__ __align__(16) unsigned short Bs[GBN * BPAD];   // [n][k] (transposed)

  const int tid  = threadIdx.x;
  const int wave = tid >> 5;
  const int lane = tid & 31;
  const int half = lane >> 4;
  const int l15  = lane & 15;
  const int wr   = wave >> 1;   // 16-row sub-tile
  const int wc   = wave & 1;    // 32-col sub-tile

  const int mBase = blockIdx.y * GBM;
  const int nBase = blockIdx.x * GBN;

  // uniform bases (SGPRs) + 32-bit per-lane offsets
  const float* __restrict__ Ab = A + (size_t)mBase * lda;   // + (row*lda + k + k0)
  const float* __restrict__ Bb = B + nBase;                 // + ((k0+bk)*ldb + bn)

  // A staging: 512 float2 slots; thread handles 4. row = s>>4, kpair = (s&15)*2
  const int aRow0 = tid >> 4;          // + 8*i
  const int aK    = (tid & 15) * 2;    // constant
  // B staging: 1024 k-pair slots; thread handles 8. bn = s&63, bk = (s>>6)*2
  const int bN    = tid & 63;          // constant
  const int bK0   = (tid >> 6) * 2;    // + 4*i
  const int bCol  = nBase + bN;
  const bool bIn  = (bCol < N);

  v8f acc0 = {}; v8f acc1 = {};

  for (int k0 = 0; k0 < K; k0 += GBK) {
    // stage A tile (32x32): float2 loads along k, packed b32 LDS stores
    #pragma unroll
    for (int i = 0; i < 4; ++i) {
      int row = aRow0 + 8 * i;
      int off = row * lda + aK + k0;               // 32-bit
      float2 v = *(const float2*)(Ab + off);
      *(unsigned int*)&As[row * APAD + aK] = f2bf2(v.x, v.y);
    }
    // stage B tile (32x64): scalar loads (ldb may be odd), packed b32 stores
    #pragma unroll
    for (int i = 0; i < 8; ++i) {
      int bk = bK0 + 4 * i;
      int off = (k0 + bk) * ldb + bN;              // 32-bit
      float v0 = bIn ? Bb[off]       : 0.0f;
      float v1 = bIn ? Bb[off + ldb] : 0.0f;
      *(unsigned int*)&Bs[bN * BPAD + bk] = f2bf2(v0, v1);
    }
    __syncthreads();

    // A fragment: lane(half,m) holds K = {8h..8h+7} and {16+8h..16+8h+7}
    const uint4* rowA = (const uint4*)&As[(wr * 16 + l15) * APAD];
    U32x8 af; af.lo = rowA[half]; af.hi = rowA[2 + half];
    v16bf av = __builtin_bit_cast(v16bf, af);

    // B fragments: lane(half,n) holds K = {8h..8h+7} and {16+8h..16+8h+7}, column n
    const uint4* rowB0 = (const uint4*)&Bs[(wc * 32 + l15) * BPAD];
    U32x8 b0; b0.lo = rowB0[half]; b0.hi = rowB0[2 + half];
    v16bf bv0 = __builtin_bit_cast(v16bf, b0);
    const uint4* rowB1 = (const uint4*)&Bs[(wc * 32 + 16 + l15) * BPAD];
    U32x8 b1; b1.lo = rowB1[half]; b1.hi = rowB1[2 + half];
    v16bf bv1 = __builtin_bit_cast(v16bf, b1);

    acc0 = __builtin_amdgcn_wmma_f32_16x16x32_bf16(false, av, false, bv0, (short)0, acc0, false, false);
    acc1 = __builtin_amdgcn_wmma_f32_16x16x32_bf16(false, av, false, bv1, (short)0, acc1, false, false);
    __syncthreads();
  }

  // store: VGPR r -> M = r + 8*half, N = lane&15
  #pragma unroll
  for (int r = 0; r < 8; ++r) {
    int grow = mBase + wr * 16 + r + 8 * half;
    long long ro = (long long)(grow % rows_mod) * stride_in +
                   (long long)(grow / rows_mod) * stride_out;
    int col0 = nBase + wc * 32 + l15;
    int col1 = col0 + 16;
    if (col0 < N) {
      float v = acc0[r] + (bias ? bias[col0] : 0.0f);
      Out[ro + col0] = accum ? (Out[ro + col0] + v) : v;
    }
    if (col1 < N) {
      float v = acc1[r] + (bias ? bias[col1] : 0.0f);
      Out[ro + col1] = accum ? (Out[ro + col1] + v) : v;
    }
  }
}

// ---------------------------------------------------------------------------
// emb (b,l,e) -> emb_t rows ordered (t*32+b); also fills cat[:, 2H:2H+E]
// ---------------------------------------------------------------------------
__global__ __launch_bounds__(128) void prep_kernel(
    const float* __restrict__ emb, float* __restrict__ emb_t, float* __restrict__ cat)
{
  int row = blockIdx.x;              // t*32 + b
  int b = row & 31, t = row >> 5;
  const float* src = emb + ((size_t)b * LDIM + t) * EDIM;
  float* d1 = emb_t + (size_t)row * EDIM;
  float* d2 = cat   + (size_t)row * CATD + 2 * HDIM;
  for (int i = threadIdx.x; i < EDIM; i += 128) { float v = src[i]; d1[i] = v; d2[i] = v; }
}

// ---------------------------------------------------------------------------
// attention scores + softmax; writes att (ws) and atts output slice
// ---------------------------------------------------------------------------
__global__ __launch_bounds__(256) void att_kernel(
    const float* __restrict__ h, const float* __restrict__ enc,
    const unsigned char* __restrict__ mask,
    float* __restrict__ att, float* __restrict__ atts_out, int t)
{
  __shared__ float hb[HDIM];
  __shared__ float sc[NDIM];
  __shared__ float red[8];
  int b = blockIdx.x, tid = threadIdx.x;
  int wave = tid >> 5, lane = tid & 31;

  for (int i = tid; i < HDIM; i += 256) hb[i] = h[b * HDIM + i];
  __syncthreads();

  for (int n = wave; n < NDIM; n += 8) {
    const float* ep = enc + ((size_t)b * NDIM + n) * HDIM;
    float a = 0.0f;
    for (int k = lane; k < HDIM; k += 32) a += hb[k] * ep[k];
    for (int off = 16; off; off >>= 1) a += __shfl_xor(a, off, 32);
    if (lane == 0) sc[n] = mask[b * NDIM + n] ? a * (1.0f / 32.0f) : -1.0e9f;
  }
  __syncthreads();

  float m = -3.4e38f;
  for (int i = tid; i < NDIM; i += 256) m = fmaxf(m, sc[i]);
  for (int off = 16; off; off >>= 1) m = fmaxf(m, __shfl_xor(m, off, 32));
  if (lane == 0) red[wave] = m;
  __syncthreads();
  float gm = red[0];
  for (int i = 1; i < 8; ++i) gm = fmaxf(gm, red[i]);
  __syncthreads();

  float s = 0.0f;
  for (int i = tid; i < NDIM; i += 256) { float e = expf(sc[i] - gm); sc[i] = e; s += e; }
  for (int off = 16; off; off >>= 1) s += __shfl_xor(s, off, 32);
  if (lane == 0) red[wave] = s;
  __syncthreads();
  float gs = 0.0f;
  for (int i = 0; i < 8; ++i) gs += red[i];
  float inv = 1.0f / gs;

  for (int i = tid; i < NDIM; i += 256) {
    float a = sc[i] * inv;
    att[b * NDIM + i] = a;
    atts_out[((size_t)b * LDIM + t) * NDIM + i] = a;
  }
}

// context: c[b][k] = sum_n att[b][n] * enc[b][n][k]
__global__ __launch_bounds__(256) void ctx_kernel(
    const float* __restrict__ att, const float* __restrict__ enc, float* __restrict__ cvec)
{
  __shared__ float ab[NDIM];
  int b = blockIdx.x, tid = threadIdx.x;
  for (int i = tid; i < NDIM; i += 256) ab[i] = att[b * NDIM + i];
  __syncthreads();
  for (int k = tid; k < HDIM; k += 256) {
    const float* ep = enc + (size_t)b * NDIM * HDIM + k;
    float a = 0.0f;
    for (int n = 0; n < NDIM; ++n) a += ab[n] * ep[(size_t)n * HDIM];
    cvec[b * HDIM + k] = a;
  }
}

// r,u gates; hr = h * r
__global__ __launch_bounds__(256) void gate_kernel(
    const float* __restrict__ Xr, const float* __restrict__ Xu,
    const float* __restrict__ tr, const float* __restrict__ tu,
    const float* __restrict__ h, float* __restrict__ hr, float* __restrict__ ubuf, int t)
{
  int idx = blockIdx.x * 256 + threadIdx.x;   // 0 .. 32*1024-1
  int b = idx >> 10, j = idx & 1023;
  int row = t * BDIM + b;
  float r = sigf(Xr[(size_t)row * HDIM + j] + tr[idx]);
  float u = sigf(Xu[(size_t)row * HDIM + j] + tu[idx]);
  hr[idx] = h[idx] * r;
  ubuf[idx] = u;
}

// s = tanh(...); h_new; fill cat[:,0:H] and cat[:,H:2H]
__global__ __launch_bounds__(256) void update_kernel(
    const float* __restrict__ Xw, const float* __restrict__ ts,
    const float* __restrict__ ubuf, const float* __restrict__ cvec,
    float* __restrict__ h, float* __restrict__ cat, int t)
{
  int idx = blockIdx.x * 256 + threadIdx.x;
  int b = idx >> 10, j = idx & 1023;
  int row = t * BDIM + b;
  float s = tanhf(Xw[(size_t)row * HDIM + j] + ts[idx]);
  float u = ubuf[idx];
  float hn = (1.0f - u) * h[idx] + u * s;
  h[idx] = hn;
  cat[(size_t)row * CATD + j] = hn;
  cat[(size_t)row * CATD + HDIM + j] = cvec[idx];
}

// p_gen[b][t] = sigmoid(cat[row] . pg_w + pg_b)
__global__ __launch_bounds__(256) void pgen_kernel(
    const float* __restrict__ cat, const float* __restrict__ pgw,
    const float* __restrict__ pgb, float* __restrict__ out)
{
  __shared__ float red[8];
  int row = blockIdx.x, tid = threadIdx.x;
  const float* cp = cat + (size_t)row * CATD;
  float a = 0.0f;
  for (int i = tid; i < CATD; i += 256) a += cp[i] * pgw[i];
  for (int off = 16; off; off >>= 1) a += __shfl_xor(a, off, 32);
  if ((tid & 31) == 0) red[tid >> 5] = a;
  __syncthreads();
  if (tid == 0) {
    float s = 0.0f;
    for (int i = 0; i < 8; ++i) s += red[i];
    int b = row & 31, t = row >> 5;
    out[b * LDIM + t] = sigf(s + pgb[0]);
  }
}

// ---------------------------------------------------------------------------
extern "C" void kernel_launch(void* const* d_in, const int* in_sizes, int n_in,
                              void* d_out, int out_size, void* d_ws, size_t ws_size,
                              hipStream_t stream)
{
  (void)in_sizes; (void)n_in; (void)out_size; (void)ws_size;
  const float* emb     = (const float*)d_in[0];
  const float* hidden0 = (const float*)d_in[1];
  const float* enc     = (const float*)d_in[2];
  const unsigned char* mask = (const unsigned char*)d_in[3];
  const float* Wr = (const float*)d_in[4];
  const float* Wu = (const float*)d_in[5];
  const float* Ur = (const float*)d_in[6];
  const float* Uu = (const float*)d_in[7];
  const float* Cr = (const float*)d_in[8];
  const float* Cu = (const float*)d_in[9];
  const float* Wm = (const float*)d_in[10];
  const float* Um = (const float*)d_in[11];
  const float* Cm = (const float*)d_in[12];
  const float* fc_w = (const float*)d_in[13];
  const float* fc_b = (const float*)d_in[14];
  const float* pg_w = (const float*)d_in[15];
  const float* pg_b = (const float*)d_in[16];

  float* out = (float*)d_out;
  const long long O_H   = (long long)BDIM * LDIM * VDIM;     // hidden_final
  const long long O_ATT = O_H + (long long)BDIM * HDIM;      // atts
  const long long O_PG  = O_ATT + (long long)BDIM * LDIM * NDIM;

  const int ROWS = BDIM * LDIM;     // 1024
  float* ws = (float*)d_ws;
  float* emb_t = ws; ws += (size_t)ROWS * EDIM;
  float* Xr   = ws; ws += (size_t)ROWS * HDIM;
  float* Xu   = ws; ws += (size_t)ROWS * HDIM;
  float* Xw   = ws; ws += (size_t)ROWS * HDIM;
  float* hbuf = ws; ws += (size_t)BDIM * HDIM;
  float* attb = ws; ws += (size_t)BDIM * NDIM;
  float* cvec = ws; ws += (size_t)BDIM * HDIM;
  float* tr   = ws; ws += (size_t)BDIM * HDIM;
  float* tu   = ws; ws += (size_t)BDIM * HDIM;
  float* tsb  = ws; ws += (size_t)BDIM * HDIM;
  float* hrb  = ws; ws += (size_t)BDIM * HDIM;
  float* ubuf = ws; ws += (size_t)BDIM * HDIM;
  float* cat  = ws; ws += (size_t)ROWS * CATD;

  auto gemm = [&](const float* A, int lda, const float* B, int ldb, const float* bias,
                  float* O, int M, int N, int K,
                  int rows_mod, long long sin, long long sout, int accum) {
    dim3 g((N + GBN - 1) / GBN, M / GBM);
    gemm_bf16_wmma<<<g, 128, 0, stream>>>(A, lda, B, ldb, bias, O, M, N, K,
                                          rows_mod, sin, sout, accum);
  };

  // h <- initial hidden
  hipMemcpyAsync(hbuf, hidden0, (size_t)BDIM * HDIM * sizeof(float),
                 hipMemcpyDeviceToDevice, stream);

  // step-independent: emb_t (row = t*32+b) and cat w-slice
  prep_kernel<<<ROWS, 128, 0, stream>>>(emb, emb_t, cat);

  // X* = emb_t @ {Wr,Wu,W}  (1024 x 1024 each), read weights once
  gemm(emb_t, EDIM, Wr, HDIM, nullptr, Xr, ROWS, HDIM, EDIM, ROWS, HDIM, 0, 0);
  gemm(emb_t, EDIM, Wu, HDIM, nullptr, Xu, ROWS, HDIM, EDIM, ROWS, HDIM, 0, 0);
  gemm(emb_t, EDIM, Wm, HDIM, nullptr, Xw, ROWS, HDIM, EDIM, ROWS, HDIM, 0, 0);

  // sequential GRU recurrence (U/C weights: 24 MB, L2-resident across steps)
  for (int t = 0; t < LDIM; ++t) {
    att_kernel<<<BDIM, 256, 0, stream>>>(hbuf, enc, mask, attb, out + O_ATT, t);
    ctx_kernel<<<BDIM, 256, 0, stream>>>(attb, enc, cvec);

    gemm(hbuf, HDIM, Ur, HDIM, nullptr, tr, BDIM, HDIM, HDIM, BDIM, HDIM, 0, 0);
    gemm(cvec, HDIM, Cr, HDIM, nullptr, tr, BDIM, HDIM, HDIM, BDIM, HDIM, 0, 1);
    gemm(hbuf, HDIM, Uu, HDIM, nullptr, tu, BDIM, HDIM, HDIM, BDIM, HDIM, 0, 0);
    gemm(cvec, HDIM, Cu, HDIM, nullptr, tu, BDIM, HDIM, HDIM, BDIM, HDIM, 0, 1);

    gate_kernel<<<(BDIM * HDIM) / 256, 256, 0, stream>>>(Xr, Xu, tr, tu, hbuf, hrb, ubuf, t);

    gemm(hrb, HDIM, Um, HDIM, nullptr, tsb, BDIM, HDIM, HDIM, BDIM, HDIM, 0, 0);
    gemm(cvec, HDIM, Cm, HDIM, nullptr, tsb, BDIM, HDIM, HDIM, BDIM, HDIM, 0, 1);

    update_kernel<<<(BDIM * HDIM) / 256, 256, 0, stream>>>(Xw, tsb, ubuf, cvec, hbuf, cat, t);
  }

  // logits: ONE pass over fc_w (514 MB read once instead of 32x).
  // row r = t*32+b maps to out[(b*L + t)*V]: rows_mod=32, sin=L*V, sout=V.
  gemm(cat, CATD, fc_w, VDIM, fc_b, out, ROWS, VDIM, CATD,
       BDIM, (long long)LDIM * VDIM, (long long)VDIM, 0);

  pgen_kernel<<<ROWS, 256, 0, stream>>>(cat, pg_w, pg_b, out + O_PG);

  // hidden_final
  hipMemcpyAsync(out + O_H, hbuf, (size_t)BDIM * HDIM * sizeof(float),
                 hipMemcpyDeviceToDevice, stream);
}
